// Model1_39676907885945
// MI455X (gfx1250) — matrix operand order repaired and checked by reading, not compiled, not run on hardware
//
#include <hip/hip_runtime.h>
#include <stdint.h>

#define Bsz 4
#define Hn  16
#define Ssz 2048
#define Dd  64
#define QT  128   // q rows per block (8 waves x 16)
#define KT  64    // keys per tile
#define NW  8
#define LOG2E 1.4426950408889634f

#if defined(__gfx1250__) && __has_builtin(__builtin_amdgcn_tensor_load_to_lds)
#define HAVE_TDM 1
#else
#define HAVE_TDM 0
#endif

typedef __attribute__((ext_vector_type(16))) __bf16 v16bf;
typedef __attribute__((ext_vector_type(8)))  __bf16 v8bf;
typedef __attribute__((ext_vector_type(4)))  __bf16 v4bf;
typedef __attribute__((ext_vector_type(8)))  float  v8f;
typedef __attribute__((ext_vector_type(4)))  unsigned int ui32x4;
typedef __attribute__((ext_vector_type(8)))  int i32x8;
typedef __attribute__((ext_vector_type(4)))  int i32x4;

union BF16Frag { v16bf v; v8bf h[2]; };

__device__ __forceinline__ float rowmax16(float x) {
#pragma unroll
    for (int m = 8; m; m >>= 1) x = fmaxf(x, __shfl_xor(x, m, 16));
    return x;
}
__device__ __forceinline__ float rowsum16(float x) {
#pragma unroll
    for (int m = 8; m; m >>= 1) x += __shfl_xor(x, m, 16);
    return x;
}

#if HAVE_TDM
// Issue a TDM 2D tile load: 64x64 f32 tile from a [Ssz x Dd] f32 tensor -> LDS.
// D# per CDNA5 ISA ch.8: group0 = {count, lds_addr, global_addr, type=2},
// group1 = {data_size=4B, tensor_dim0=Dd, tensor_dim1=Ssz, tile 64x64, stride0=Dd}.
// This toolchain's builtin takes 6 args (g0, g1, g2, g3, g4, cpol).
__device__ __forceinline__ void tdm_load_tile(const float* gaddr, unsigned lds_byte_off) {
    unsigned long long ga = (unsigned long long)(uintptr_t)gaddr;
    ui32x4 g0;
    g0[0] = 1u;                                                 // count=1, user D#
    g0[1] = lds_byte_off;                                       // lds_addr [63:32]
    g0[2] = (unsigned)(ga & 0xffffffffu);                       // global_addr [95:64]
    g0[3] = (unsigned)((ga >> 32) & 0x01ffffffu) | (2u << 30);  // addr[56:32] | type=2

    i32x8 g1;
    g1[0] = 0x00020000;                  // workgroup_mask=0, data_size=2 (4 bytes)
    g1[1] = (int)((unsigned)Dd << 16);   // tensor_dim0[15:0] in bits[63:48]
    g1[2] = (int)((unsigned)Ssz << 16);  // tensor_dim0 hi=0, tensor_dim1[15:0]
    g1[3] = (int)((unsigned)KT << 16);   // tensor_dim1 hi=0, tile_dim0=64
    g1[4] = KT;                          // tile_dim1=64, tile_dim2=0
    g1[5] = Dd;                          // tensor_dim0_stride[31:0] = 64
    g1[6] = 0;
    g1[7] = 0;

    i32x4 z4 = { 0, 0, 0, 0 };           // groups 2/3 unused (<=2D tensor)
    i32x8 z8 = { 0, 0, 0, 0, 0, 0, 0, 0 };
    __builtin_amdgcn_tensor_load_to_lds(g0, g1, z4, z4, z8, 0);
}
#endif

__global__ __launch_bounds__(256)
void flash_attn_scaled(const float* __restrict__ Q, const float* __restrict__ K,
                       const float* __restrict__ V, const float* __restrict__ SF,
                       const float* __restrict__ ISC, float* __restrict__ O)
{
    __shared__ __align__(16) __bf16 sK[Dd][KT];       // K tile natural   [key][d]    8 KB
    __shared__ __align__(16) __bf16 sV[KT][Dd];       // V tile transposed [d][key]   8 KB
    __shared__ __align__(16) __bf16 sP[NW][16][KT];   // per-wave P strip / Q staging 16 KB
#if HAVE_TDM
    __shared__ __align__(16) float rawK[KT * Dd];     // TDM landing buffers (f32)    16 KB
    __shared__ __align__(16) float rawV[KT * Dd];     //                              16 KB
#endif

    const int tid  = threadIdx.x;
    const int wave = tid >> 5;
    const int lane = tid & 31;
    const int half = lane >> 4;
    const int l16  = lane & 15;

    const int bh = blockIdx.y;
    const int q0 = blockIdx.x * QT;
    const int qw = q0 + wave * 16;          // wave's first q row

    const float* Qb = Q + (size_t)bh * Ssz * Dd;
    const float* Kb = K + (size_t)bh * Ssz * Dd;
    const float* Vb = V + (size_t)bh * Ssz * Dd;
    const float inv_is = 1.0f / ISC[bh];

#if HAVE_TDM
    // Kick off DMA of the first K/V tile immediately (overlaps Q staging below).
    if (wave == 0) {
        tdm_load_tile(Kb, (unsigned)(uintptr_t)&rawK[0]);
        tdm_load_tile(Vb, (unsigned)(uintptr_t)&rawV[0]);
    }
#endif

    // ---- Stage Q tile (QT x Dd) as bf16 into sP (exactly 128*64 elements) ----
    {
        __bf16* qs = &sP[0][0][0];
#pragma unroll
        for (int it = 0; it < (QT * Dd) / (256 * 4); ++it) {
            int i4 = tid + it * 256;
            float4 f = *(const float4*)(Qb + (size_t)q0 * Dd + (size_t)i4 * 4);
            v4bf p = { (__bf16)f.x, (__bf16)f.y, (__bf16)f.z, (__bf16)f.w };
            *(v4bf*)(qs + i4 * 4) = p;
        }
    }
    __syncthreads();

    // ---- Each wave loads its Q A-fragments (ISA 16-bit A 16x32 layout) ----
    v16bf qA[2];
    {
        const __bf16* qs = &sP[0][0][0];
        const int m = wave * 16 + l16;
#pragma unroll
        for (int f = 0; f < 2; ++f) {
            BF16Frag u;
            const __bf16* p = qs + m * Dd + f * 32 + 8 * half;
            u.h[0] = *(const v8bf*)(p);
            u.h[1] = *(const v8bf*)(p + 16);
            qA[f] = u.v;
        }
    }

    // ---- Per-row combined scale; C-fragment rows owned by lane: r(i)=8*half+i ----
    float rscale[8];
#pragma unroll
    for (int i = 0; i < 8; ++i) {
        int row = qw + 8 * half + i;
        rscale[i] = inv_is / SF[(size_t)bh * Ssz + row];
    }

    float mrow[8], lrow[8];
    v8f acc[4];
#pragma unroll
    for (int i = 0; i < 8; ++i) { mrow[i] = -1e30f; lrow[i] = 0.0f; }
#pragma unroll
    for (int dt = 0; dt < 4; ++dt) { v8f z = {}; acc[dt] = z; }

    // ================= main loop over key tiles =================
    for (int kt = 0; kt < Ssz; kt += KT) {
#if HAVE_TDM
        if (wave == 0) __builtin_amdgcn_s_wait_tensorcnt(0);   // tile arrived
        __syncthreads();   // publish raw tile; prior compute's sK/sV reads done

        // ---- convert raw f32 (LDS) -> bf16 sK / sV (transposed) ----
#pragma unroll
        for (int it = 0; it < (KT * Dd) / (256 * 4); ++it) {
            int i4  = tid + it * 256;
            int e   = i4 * 4;
            int key = e >> 6;
            int d   = e & 63;
            float4 kf = *(const float4*)(&rawK[e]);
            float4 vf = *(const float4*)(&rawV[e]);
            sK[key][d + 0] = (__bf16)kf.x;
            sK[key][d + 1] = (__bf16)kf.y;
            sK[key][d + 2] = (__bf16)kf.z;
            sK[key][d + 3] = (__bf16)kf.w;
            sV[d + 0][key] = (__bf16)vf.x;
            sV[d + 1][key] = (__bf16)vf.y;
            sV[d + 2][key] = (__bf16)vf.z;
            sV[d + 3][key] = (__bf16)vf.w;
        }
        __syncthreads();   // raw free again + bf16 tiles ready

        // DMA next tile while we compute on this one
        if (wave == 0 && kt + KT < Ssz) {
            tdm_load_tile(Kb + (size_t)(kt + KT) * Dd, (unsigned)(uintptr_t)&rawK[0]);
            tdm_load_tile(Vb + (size_t)(kt + KT) * Dd, (unsigned)(uintptr_t)&rawV[0]);
        }
#else
        // ---- fallback: direct global staging (round-1 path) ----
#pragma unroll
        for (int it = 0; it < (KT * Dd) / (256 * 4); ++it) {
            int i4  = tid + it * 256;
            int e   = i4 * 4;
            int key = e >> 6;
            int d   = e & 63;
            float4 kf = *(const float4*)(Kb + (size_t)kt * Dd + e);
            float4 vf = *(const float4*)(Vb + (size_t)kt * Dd + e);
            sK[key][d + 0] = (__bf16)kf.x;
            sK[key][d + 1] = (__bf16)kf.y;
            sK[key][d + 2] = (__bf16)kf.z;
            sK[key][d + 3] = (__bf16)kf.w;
            sV[d + 0][key] = (__bf16)vf.x;
            sV[d + 1][key] = (__bf16)vf.y;
            sV[d + 2][key] = (__bf16)vf.z;
            sV[d + 3][key] = (__bf16)vf.w;
        }
        if (kt + KT < Ssz) {
            __builtin_prefetch((const void*)(Kb + (size_t)(kt + KT) * Dd + tid * 16), 0, 1);
            __builtin_prefetch((const void*)(Vb + (size_t)(kt + KT) * Dd + tid * 16), 0, 1);
        }
        __syncthreads();
#endif

        // ---- GEMM1: S(16x64) = Qstrip(16x64) x K^T ----
        v8f st[4];
        {
            v8f zc = {};
#pragma unroll
            for (int ct = 0; ct < 4; ++ct) {
                BF16Frag b0, b1;   // B col = key (l16), elements = d in table order
                const __bf16* p0 = &sK[ct * 16 + l16][8 * half];
                const __bf16* p1 = &sK[ct * 16 + l16][32 + 8 * half];
                b0.h[0] = *(const v8bf*)p0; b0.h[1] = *(const v8bf*)(p0 + 16);
                b1.h[0] = *(const v8bf*)p1; b1.h[1] = *(const v8bf*)(p1 + 16);
                v8f c = __builtin_amdgcn_wmma_f32_16x16x32_bf16(
                            false, qA[0], false, b0.v, (short)0, zc, false, false);
                c     = __builtin_amdgcn_wmma_f32_16x16x32_bf16(
                            false, qA[1], false, b1.v, (short)0, c,  false, false);
                st[ct] = c;
            }
        }

        // ---- online softmax (rows live in 16-lane groups) ----
#pragma unroll
        for (int i = 0; i < 8; ++i) {
            float s0 = st[0][i] * rscale[i];
            float s1 = st[1][i] * rscale[i];
            float s2 = st[2][i] * rscale[i];
            float s3 = st[3][i] * rscale[i];
            float mx = rowmax16(fmaxf(fmaxf(s0, s1), fmaxf(s2, s3)));
            float nm = fmaxf(mrow[i], mx);
            float corr = __builtin_amdgcn_exp2f((mrow[i] - nm) * LOG2E);
            mrow[i] = nm;
            float e0 = __builtin_amdgcn_exp2f((s0 - nm) * LOG2E);
            float e1 = __builtin_amdgcn_exp2f((s1 - nm) * LOG2E);
            float e2 = __builtin_amdgcn_exp2f((s2 - nm) * LOG2E);
            float e3 = __builtin_amdgcn_exp2f((s3 - nm) * LOG2E);
            lrow[i] = lrow[i] * corr + rowsum16(e0 + e1 + e2 + e3);
#pragma unroll
            for (int dt = 0; dt < 4; ++dt) acc[dt][i] *= corr;
            const int r = 8 * half + i;
            sP[wave][r][ 0 + l16] = (__bf16)e0;
            sP[wave][r][16 + l16] = (__bf16)e1;
            sP[wave][r][32 + l16] = (__bf16)e2;
            sP[wave][r][48 + l16] = (__bf16)e3;
        }

        // ---- reload P as A-fragments (own strip, no cross-wave barrier needed) ----
        v16bf pA[2];
#pragma unroll
        for (int f = 0; f < 2; ++f) {
            BF16Frag u;
            const __bf16* p = &sP[wave][l16][f * 32 + 8 * half];
            u.h[0] = *(const v8bf*)(p);
            u.h[1] = *(const v8bf*)(p + 16);
            pA[f] = u.v;
        }

        // ---- GEMM2: O(16x64) += P(16x64) x V(64x64) ----
#pragma unroll
        for (int dt = 0; dt < 4; ++dt) {
            BF16Frag b0, b1;   // B col = d (l16), elements = key in table order
            const __bf16* p0 = &sV[dt * 16 + l16][8 * half];
            const __bf16* p1 = &sV[dt * 16 + l16][32 + 8 * half];
            b0.h[0] = *(const v8bf*)p0; b0.h[1] = *(const v8bf*)(p0 + 16);
            b1.h[0] = *(const v8bf*)p1; b1.h[1] = *(const v8bf*)(p1 + 16);
            acc[dt] = __builtin_amdgcn_wmma_f32_16x16x32_bf16(
                          false, pA[0], false, b0.v, (short)0, acc[dt], false, false);
            acc[dt] = __builtin_amdgcn_wmma_f32_16x16x32_bf16(
                          false, pA[1], false, b1.v, (short)0, acc[dt], false, false);
        }
#if !HAVE_TDM
        __syncthreads();   // protect sK/sV before next tile's staging
#endif
    }

    // ---- epilogue: out = acc / l ----
#pragma unroll
    for (int dt = 0; dt < 4; ++dt) {
#pragma unroll
        for (int i = 0; i < 8; ++i) {
            const int row = qw + 8 * half + i;
            O[((size_t)bh * Ssz + row) * Dd + dt * 16 + l16] = acc[dt][i] / lrow[i];
        }
    }
}

extern "C" void kernel_launch(void* const* d_in, const int* in_sizes, int n_in,
                              void* d_out, int out_size, void* d_ws, size_t ws_size,
                              hipStream_t stream) {
    const float* q   = (const float*)d_in[0];
    const float* k   = (const float*)d_in[1];
    const float* v   = (const float*)d_in[2];
    const float* sf  = (const float*)d_in[3];
    const float* isc = (const float*)d_in[4];
    float* out = (float*)d_out;

    dim3 grid(Ssz / QT, Bsz * Hn);   // 16 x 64 blocks
    flash_attn_scaled<<<grid, 256, 0, stream>>>(q, k, v, sf, isc, out);
}
